// SVTR_67345087201547
// MI455X (gfx1250) — matrix operand order repaired and checked by reading, not compiled, not run on hardware
//
#include <hip/hip_runtime.h>
#include <hip/hip_bf16.h>
#include <math.h>

// ---------------------------------------------------------------------------
// Deformable attention for MI455X (gfx1250, wave32, WMMA).
//   B=2, N=4096 (16x256 grid), D=512, H=8, P=9, HD=64.
// GEMMs: bf16 v_wmma_f32_16x16x32_bf16; weight tiles streamed via async
// global->LDS (double-buffered, ASYNCcnt pipelined); each wave computes a
// 32x64 tile (8 WMMAs per staged tile). Sampling/softmax in fp32 (the k/v
// map is 33.5 MB -> fully L2-resident on the 192 MB L2).
// ---------------------------------------------------------------------------

typedef __bf16 bf16_t;
typedef __attribute__((ext_vector_type(16))) __bf16 v16bf;
typedef __attribute__((ext_vector_type(8)))  __bf16 v8bf;
typedef __attribute__((ext_vector_type(8)))  float  v8f;

#define B_   2
#define N_   4096
#define D_   512
#define H_   8
#define P_   9
#define HD_  64
#define GH_  16
#define GW_  256
#define OFF_SCALE 4.0f
#define MROWS (B_ * N_)        // 8192
#define OFFDIM (H_ * P_ * 2)   // 144
#define OFFPAD 192             // 144 padded to a multiple of 64

union V16U { v16bf v; v8bf h[2]; };

// low 32 bits of a flat pointer into LDS == LDS byte offset (gfx1250 aperture)
__device__ __forceinline__ uint32_t lds_addr32(const void* p) {
  return (uint32_t)(uintptr_t)p;
}

// one async 16B transfer per lane: global -> LDS, tracked by ASYNCcnt
__device__ __forceinline__ void async_load_b128(uint32_t lds_off,
                                                const void* gaddr) {
  asm volatile("global_load_async_to_lds_b128 %0, %1, off"
               :: "v"(lds_off), "v"((uint64_t)(uintptr_t)gaddr)
               : "memory");
}

// ------------------------ f32 -> bf16 (with zero pad) ----------------------
__global__ void cvt_pad_f32_bf16(const float* __restrict__ in,
                                 bf16_t* __restrict__ out,
                                 int in_count, int out_count) {
  int i = blockIdx.x * blockDim.x + threadIdx.x;
  int stride = gridDim.x * blockDim.x;
  for (; i < out_count; i += stride)
    out[i] = (i < in_count) ? (bf16_t)in[i] : (bf16_t)0.0f;
}

__global__ void pad_f32(const float* __restrict__ in, float* __restrict__ out,
                        int in_count, int out_count) {
  int i = blockIdx.x * blockDim.x + threadIdx.x;
  if (i < out_count) out[i] = (i < in_count) ? in[i] : 0.0f;
}

// ------------------------------- WMMA GEMM ---------------------------------
// C[m,n] = act( sum_k A[m,k] * W[n,k] + bias[n] ), Nout % 64 == 0, K % 32 == 0.
// Block: 256 thr = 8 waves; each wave owns 32 rows x 64 cols -> 256 rows/block.
// grid = (Nout/64, M/256). The 64x32 weight tile (4KB) is staged once per
// block in LDS in exact B-fragment order (async global->LDS, double-buffered);
// all 4 B fragments are preloaded, then 8 back-to-back WMMAs consume them.
template <int ACT>
__global__ __launch_bounds__(256) void gemm_bf16_wmma(
    const bf16_t* __restrict__ A, const bf16_t* __restrict__ W,
    const float* __restrict__ bias, float* __restrict__ C,
    int M, int K, int Nout) {
  __shared__ __align__(16) bf16_t smem[2][64 * 32];  // 2 x 4KB

  const int tid  = threadIdx.x;
  const int lane = tid & 31;
  const int wave = tid >> 5;
  const int l15  = lane & 15;
  const int hsel = lane >> 4;          // 0: lanes 0-15, 1: lanes 16-31
  const int m0   = blockIdx.y * 256 + wave * 32;
  const int n0   = blockIdx.x * 64;

  // staging map: entry e = (tile bt, frag-lane bl) covers 32B; half hh = 16B
  const int e  = tid >> 1, hh = tid & 1;
  const int bt = e >> 5,  bl = e & 31;
  // B-fragment layout: frag-lane bl holds column n0+bt*16+(bl&15),
  // K {(bl>>4)*8 + 0..7, 16 + (bl>>4)*8 + 0..7} (matches A layout)
  const bf16_t* __restrict__ gb =
      W + (size_t)(n0 + bt * 16 + (bl & 15)) * K + (bl >> 4) * 8 + hh * 16;
  const uint32_t ldsw0 = lds_addr32(&smem[0][0]) + (uint32_t)(e * 32 + hh * 16);
  const uint32_t ldsw1 = lds_addr32(&smem[1][0]) + (uint32_t)(e * 32 + hh * 16);

  v8f acc[2][4];
#pragma unroll
  for (int mt = 0; mt < 2; ++mt)
#pragma unroll
    for (int t = 0; t < 4; ++t)
#pragma unroll
      for (int r = 0; r < 8; ++r) acc[mt][t][r] = 0.0f;

  // A-fragment: lane l15 holds row; half 0: K {0..7,16..23},
  // half 1: K {8..15,24..31} (CDNA5 16-bit A layout)
  const bf16_t* __restrict__ arow0 = A + (size_t)(m0 + l15) * K + hsel * 8;
  const bf16_t* __restrict__ arow1 = A + (size_t)(m0 + 16 + l15) * K + hsel * 8;

  const int S = K >> 5;                 // K-steps of 32
  async_load_b128(ldsw0, gb);           // prologue: fill buffer 0

  // steady state: always one tile in flight, no branches around the asm
  for (int s = 0; s < S - 1; ++s) {
    const int k0 = s << 5;

    V16U af0, af1;
    af0.h[0] = *(const v8bf*)(arow0 + k0);
    af0.h[1] = *(const v8bf*)(arow0 + k0 + 16);
    af1.h[0] = *(const v8bf*)(arow1 + k0);
    af1.h[1] = *(const v8bf*)(arow1 + k0 + 16);

    async_load_b128((s & 1) ? ldsw0 : ldsw1, gb + k0 + 32);
    asm volatile("s_wait_asynccnt 0x1" ::: "memory");
    __syncthreads();                    // whole 4KB tile resident

    const bf16_t* __restrict__ bufp = smem[s & 1];
    V16U bfr[4];
#pragma unroll
    for (int t = 0; t < 4; ++t) {       // preload all B fragments first
      const bf16_t* src = bufp + (t * 32 + lane) * 16;   // 32B per lane
      bfr[t].h[0] = *(const v8bf*)(src);
      bfr[t].h[1] = *(const v8bf*)(src + 8);
    }
#pragma unroll
    for (int t = 0; t < 4; ++t) {       // 8 back-to-back WMMAs
      acc[0][t] = __builtin_amdgcn_wmma_f32_16x16x32_bf16(
          false, af0.v, false, bfr[t].v, (short)0, acc[0][t], false, false);
      acc[1][t] = __builtin_amdgcn_wmma_f32_16x16x32_bf16(
          false, af1.v, false, bfr[t].v, (short)0, acc[1][t], false, false);
    }
    __syncthreads();                    // reads done before buffer reuse
  }

  // peeled final K-step
  {
    const int s  = S - 1;
    const int k0 = s << 5;
    V16U af0, af1;
    af0.h[0] = *(const v8bf*)(arow0 + k0);
    af0.h[1] = *(const v8bf*)(arow0 + k0 + 16);
    af1.h[0] = *(const v8bf*)(arow1 + k0);
    af1.h[1] = *(const v8bf*)(arow1 + k0 + 16);

    asm volatile("s_wait_asynccnt 0x0" ::: "memory");
    __syncthreads();

    const bf16_t* __restrict__ bufp = smem[s & 1];
    V16U bfr[4];
#pragma unroll
    for (int t = 0; t < 4; ++t) {
      const bf16_t* src = bufp + (t * 32 + lane) * 16;
      bfr[t].h[0] = *(const v8bf*)(src);
      bfr[t].h[1] = *(const v8bf*)(src + 8);
    }
#pragma unroll
    for (int t = 0; t < 4; ++t) {
      acc[0][t] = __builtin_amdgcn_wmma_f32_16x16x32_bf16(
          false, af0.v, false, bfr[t].v, (short)0, acc[0][t], false, false);
      acc[1][t] = __builtin_amdgcn_wmma_f32_16x16x32_bf16(
          false, af1.v, false, bfr[t].v, (short)0, acc[1][t], false, false);
    }
  }

  // Epilogue: lane l15 = column, VGPR r -> row base + hsel*8 + r (no guards:
  // Nout multiple of 64, M multiple of 256).
#pragma unroll
  for (int mt = 0; mt < 2; ++mt) {
#pragma unroll
    for (int t = 0; t < 4; ++t) {
      const int nc = n0 + t * 16 + l15;
      const float bv = bias[nc];
#pragma unroll
      for (int r = 0; r < 8; ++r) {
        const int mr = m0 + mt * 16 + hsel * 8 + r;
        float val = acc[mt][t][r] + bv;
        if (ACT == 1) val = tanhf(val) * OFF_SCALE;
        C[(size_t)mr * Nout + nc] = val;
      }
    }
  }
}

// --------------------- deformable sampling + attention ---------------------
// One wave per (b,h,n). hd=64 -> 2 channels per lane (wave32).
// q:    (B*N, 512) f32    offs: (B*N, OFFPAD) f32 (tanh*4 applied)
// kv:   (B*N, 1024) f32   out:  (B*N, 512) bf16 (feeds final WMMA GEMM)
__global__ __launch_bounds__(256) void deform_attn_kernel(
    const float* __restrict__ q, const float* __restrict__ offs,
    const float* __restrict__ kv, bf16_t* __restrict__ out) {
  const int wid  = blockIdx.x * (blockDim.x >> 5) + (threadIdx.x >> 5);
  const int lane = threadIdx.x & 31;
  const int n = wid & (N_ - 1);
  const int h = (wid >> 12) & (H_ - 1);
  const int b = wid >> 15;

  const int c0  = lane * 2;
  const int row = b * N_ + n;

  const float* __restrict__ qp = q + (size_t)row * D_ + h * HD_;
  const float q0 = qp[c0];
  const float q1 = qp[c0 + 1];

  const float basex = (float)(n & (GW_ - 1));
  const float basey = (float)(n >> 8);

  float sv0[P_], sv1[P_], lg[P_];

  const float* __restrict__ op =
      offs + (size_t)row * OFFPAD + (size_t)h * (P_ * 2);

#pragma unroll
  for (int p = 0; p < P_; ++p) {
    const float sx = basex + op[p * 2 + 0];
    const float sy = basey + op[p * 2 + 1];
    const float x0f = floorf(sx), y0f = floorf(sy);
    const float wx = sx - x0f, wy = sy - y0f;
    const int x0 = (int)x0f, y0 = (int)y0f;

    float k0 = 0.f, k1 = 0.f, v0 = 0.f, v1 = 0.f;

    const int   xs[4] = {x0, x0 + 1, x0, x0 + 1};
    const int   ys[4] = {y0, y0, y0 + 1, y0 + 1};
    const float ws[4] = {(1.f - wx) * (1.f - wy), wx * (1.f - wy),
                         (1.f - wx) * wy,         wx * wy};
#pragma unroll
    for (int cnr = 0; cnr < 4; ++cnr) {
      const int xi = xs[cnr], yi = ys[cnr];
      if (xi >= 0 && xi < GW_ && yi >= 0 && yi < GH_) {
        const int tok = yi * GW_ + xi;
        const float* __restrict__ kp =
            kv + (size_t)(b * N_ + tok) * (2 * D_) + h * HD_;
        const float wt = ws[cnr];
        k0 += wt * kp[c0];
        k1 += wt * kp[c0 + 1];
        v0 += wt * kp[D_ + c0];
        v1 += wt * kp[D_ + c0 + 1];
      }
    }

    // wave32 butterfly reduction of the 64-wide q.k dot product
    float s = q0 * k0 + q1 * k1;
#pragma unroll
    for (int m = 16; m >= 1; m >>= 1) s += __shfl_xor(s, m, 32);

    lg[p]  = s * 0.125f;   // HD^-0.5 = 1/8
    sv0[p] = v0;
    sv1[p] = v1;
  }

  // softmax over P=9 (every lane holds all logits after the reduction)
  float mx = lg[0];
#pragma unroll
  for (int p = 1; p < P_; ++p) mx = fmaxf(mx, lg[p]);
  float den = 0.f, ex[P_];
#pragma unroll
  for (int p = 0; p < P_; ++p) { ex[p] = expf(lg[p] - mx); den += ex[p]; }
  const float inv = 1.0f / den;

  float o0 = 0.f, o1 = 0.f;
#pragma unroll
  for (int p = 0; p < P_; ++p) {
    const float a = ex[p] * inv;
    o0 += a * sv0[p];
    o1 += a * sv1[p];
  }

  bf16_t* __restrict__ outp = out + (size_t)row * D_ + h * HD_;
  outp[c0]     = (bf16_t)o0;
  outp[c0 + 1] = (bf16_t)o1;
}

// ------------------------------ host launch --------------------------------
extern "C" void kernel_launch(void* const* d_in, const int* in_sizes, int n_in,
                              void* d_out, int out_size, void* d_ws,
                              size_t ws_size, hipStream_t stream) {
  const float* x    = (const float*)d_in[0];
  const float* Wq   = (const float*)d_in[1];
  const float* bq   = (const float*)d_in[2];
  const float* Woff = (const float*)d_in[3];
  const float* boff = (const float*)d_in[4];
  const float* Wkv  = (const float*)d_in[5];
  const float* bkv  = (const float*)d_in[6];
  const float* Wo   = (const float*)d_in[7];
  const float* bo   = (const float*)d_in[8];
  float* outp = (float*)d_out;

  // ---- workspace carving (256B aligned) ----
  char* wp = (char*)d_ws;
  auto carve = [&](size_t bytes) -> void* {
    void* p = (void*)wp;
    wp += (bytes + 255) & ~(size_t)255;
    return p;
  };
  bf16_t* x_bf     = (bf16_t*)carve((size_t)MROWS * D_ * sizeof(bf16_t));
  bf16_t* Wq_bf    = (bf16_t*)carve((size_t)D_ * D_ * sizeof(bf16_t));
  bf16_t* Woff_bf  = (bf16_t*)carve((size_t)OFFPAD * D_ * sizeof(bf16_t));
  bf16_t* Wkv_bf   = (bf16_t*)carve((size_t)(2 * D_) * D_ * sizeof(bf16_t));
  bf16_t* Wo_bf    = (bf16_t*)carve((size_t)D_ * D_ * sizeof(bf16_t));
  float*  boff_pad = (float*)carve((size_t)OFFPAD * sizeof(float));
  float*  q_f      = (float*)carve((size_t)MROWS * D_ * sizeof(float));
  float*  off_f    = (float*)carve((size_t)MROWS * OFFPAD * sizeof(float));
  float*  kv_f     = (float*)carve((size_t)MROWS * (2 * D_) * sizeof(float));
  bf16_t* attn_bf  = (bf16_t*)carve((size_t)MROWS * D_ * sizeof(bf16_t));

  // ---- stage 0: bf16 conversions (+ zero padding for the offset head) ----
  auto cvt = [&](const float* src, bf16_t* dst, int in_count, int out_count) {
    int blocks = (out_count + 255) / 256;
    if (blocks > 2048) blocks = 2048;
    cvt_pad_f32_bf16<<<blocks, 256, 0, stream>>>(src, dst, in_count, out_count);
  };
  cvt(x,    x_bf,    MROWS * D_,   MROWS * D_);
  cvt(Wq,   Wq_bf,   D_ * D_,      D_ * D_);
  cvt(Woff, Woff_bf, OFFDIM * D_,  OFFPAD * D_);
  cvt(Wkv,  Wkv_bf,  2 * D_ * D_,  2 * D_ * D_);
  cvt(Wo,   Wo_bf,   D_ * D_,      D_ * D_);
  pad_f32<<<1, OFFPAD, 0, stream>>>(boff, boff_pad, OFFDIM, OFFPAD);

  const dim3 blk(256);
  const int  my = MROWS / 256;   // 32

  // ---- stage 1: q = x @ Wq^T + bq ----
  gemm_bf16_wmma<0><<<dim3(D_ / 64, my), blk, 0, stream>>>(
      x_bf, Wq_bf, bq, q_f, MROWS, D_, D_);

  // ---- stage 2: off = tanh(x @ Woff^T + boff) * 4  (Nout padded to 192) ----
  gemm_bf16_wmma<1><<<dim3(OFFPAD / 64, my), blk, 0, stream>>>(
      x_bf, Woff_bf, boff_pad, off_f, MROWS, D_, OFFPAD);

  // ---- stage 3: kv = x @ Wkv^T + bkv ----
  gemm_bf16_wmma<0><<<dim3((2 * D_) / 64, my), blk, 0, stream>>>(
      x_bf, Wkv_bf, bkv, kv_f, MROWS, D_, 2 * D_);

  // ---- stage 4: bilinear sampling + attention (1 wave per (b,h,n)) ----
  const int total_waves = B_ * H_ * N_;          // 65536
  deform_attn_kernel<<<total_waves / 8, blk, 0, stream>>>(
      q_f, off_f, kv_f, attn_bf);

  // ---- stage 5: out = attn @ Wo^T + bo ----
  gemm_bf16_wmma<0><<<dim3(D_ / 64, my), blk, 0, stream>>>(
      attn_bf, Wo_bf, bo, outp, MROWS, D_, D_);
}
